// GFMGC_7481833030202
// MI455X (gfx1250) — matrix operand order repaired and checked by reading, not compiled, not run on hardware
//
#include <hip/hip_runtime.h>
#include <hip/hip_bf16.h>
#include <math.h>

// ---------------------------------------------------------------------------
// Types for CDNA5 WMMA (wave32, v_wmma_f32_16x16x32_bf16)
// ---------------------------------------------------------------------------
typedef __attribute__((ext_vector_type(16))) __bf16 v16bf;
typedef __attribute__((ext_vector_type(8)))  __bf16 v8bf;
typedef __attribute__((ext_vector_type(8)))  float  v8f;

// Problem constants
#define BS   64
#define SEQ  1024
#define EMB  300
#define KPAD 320          // EMB padded to multiple of 32 for bf16 WMMA
#define HH   128
#define N3H  384          // 3*H
#define NC   23

// Workspace layout (bytes)
#define WS_XM   ((size_t)0)                                  // bf16 (BS*SEQ, KPAD)
#define WS_G    (WS_XM + (size_t)BS*SEQ*KPAD*2)              // f32  [t][b][3H] time-major
#define WS_WIH  (WS_G  + (size_t)BS*SEQ*N3H*4)               // bf16 (3H, KPAD)
#define WS_WHH  (WS_WIH + (size_t)N3H*KPAD*2)                // bf16 (3H, H)
#define WS_S    (WS_WHH + (size_t)N3H*HH*2)                  // f32  (BS, EMB)  sum over t
#define WS_HF   (WS_S  + (size_t)BS*EMB*4)                   // f32  (BS, H)  fwd hidden
#define WS_HB   (WS_HF + (size_t)BS*HH*4)                    // f32  (BS, H)  bwd hidden

// ---------------------------------------------------------------------------
// Fragment helpers (16x16x32 bf16; this code's contiguous k order, same for A and B)
// A (16x32, row-major source): lane<16 row=lane holds K {kb..kb+15},
// kb=0; lanes 16..31 same rows, kb=16.
// ---------------------------------------------------------------------------
__device__ __forceinline__ v16bf load_a_frag(const __bf16* p /* row base + half*16 + kk */) {
    v8bf lo = *(const v8bf*)(p);
    v8bf hi = *(const v8bf*)(p + 8);
    v16bf a;
#pragma unroll
    for (int i = 0; i < 8; ++i) { a[i] = lo[i]; a[i + 8] = hi[i]; }
    return a;
}

__device__ __forceinline__ float sigmoidf_(float x) { return 1.0f / (1.0f + expf(-x)); }

// ---------------------------------------------------------------------------
// K0: convert forward GRU weights to bf16 (W_ih padded K 300->320)
// ---------------------------------------------------------------------------
__global__ void prep_weights(const float* __restrict__ Wih, const float* __restrict__ Whh,
                             __bf16* __restrict__ WihB, __bf16* __restrict__ WhhB) {
    int idx = blockIdx.x * blockDim.x + threadIdx.x;
    const int nih = N3H * KPAD;          // 122880
    const int nhh = N3H * HH;            // 49152
    if (idx < nih) {
        int row = idx / KPAD, col = idx % KPAD;
        float v = (col < EMB) ? Wih[row * EMB + col] : 0.0f;
        WihB[idx] = (__bf16)v;
    } else if (idx < nih + nhh) {
        int j = idx - nih;
        WhhB[j] = (__bf16)Whh[j];
    }
}

// ---------------------------------------------------------------------------
// K1: xm = x*mask -> bf16 padded rows; s[b,f] = sum_t xm   (one block per batch)
// ---------------------------------------------------------------------------
__global__ void mask_convert_sum(const float* __restrict__ x, const float* __restrict__ mask,
                                 __bf16* __restrict__ XM, float* __restrict__ S) {
    const int b = blockIdx.x;
    const int f = threadIdx.x;           // 0..KPAD-1
    float acc = 0.0f;
    for (int t = 0; t < SEQ; ++t) {
        const size_t row = (size_t)b * SEQ + t;
        float v = 0.0f;
        if (f < EMB) {
            float mk = mask[row];
            v = x[row * EMB + f] * mk;
            acc += v;
        }
        XM[row * KPAD + f] = (__bf16)v;
    }
    if (f < EMB) S[(size_t)b * EMB + f] = acc;
}

// ---------------------------------------------------------------------------
// K2: G[t][b][n] = XM[b*SEQ+t,:] @ WihB[n,:]^T + b_ih[n]
// 256 threads = 8 waves; wave -> 16 rows x 64 cols; 10 WMMA K-steps.
// ---------------------------------------------------------------------------
__global__ __launch_bounds__(256) void gemm_gi(const __bf16* __restrict__ XM,
                                               const __bf16* __restrict__ W,
                                               const float* __restrict__ bias,
                                               float* __restrict__ G) {
    const int wid  = threadIdx.x >> 5;
    const int lane = threadIdx.x & 31;
    const int half = lane >> 4;
    const int ln   = lane & 15;
    const int m0 = blockIdx.x * 128 + wid * 16;
    const int n0 = blockIdx.y * 64;

    const v8f vzero = {0.f, 0.f, 0.f, 0.f, 0.f, 0.f, 0.f, 0.f};
    v8f acc[4];
#pragma unroll
    for (int j = 0; j < 4; ++j) acc[j] = vzero;

    const __bf16* arow = XM + (size_t)(m0 + ln) * KPAD + half * 16;
#pragma unroll
    for (int kt = 0; kt < KPAD / 32; ++kt) {
        const int kk = kt * 32;
        __builtin_prefetch(arow + kk + 128, 0, 1);   // global_prefetch_b8
        v16bf a = load_a_frag(arow + kk);
#pragma unroll
        for (int j = 0; j < 4; ++j) {
            v16bf bfr = *(const v16bf*)(W + (size_t)(n0 + j * 16 + ln) * KPAD + kk + half * 16);
            acc[j] = __builtin_amdgcn_wmma_f32_16x16x32_bf16(
                false, a, false, bfr, (short)0, acc[j], false, false);
        }
    }
    // D layout: vgpr r, lane -> M = r + 8*half, N = ln.  Store time-major.
#pragma unroll
    for (int j = 0; j < 4; ++j) {
        const int n = n0 + j * 16 + ln;
        const float bv = bias[n];
#pragma unroll
        for (int r = 0; r < 8; ++r) {
            const int m  = m0 + r + 8 * half;      // flat row = b*SEQ + t
            const int t  = m & (SEQ - 1);
            const int bb = m >> 10;
            G[((size_t)t * BS + bb) * N3H + n] = acc[j][r] + bv;
        }
    }
}

// ---------------------------------------------------------------------------
// K3: sequential GRU scan. One workgroup (32 waves) on one WGP.
// Wave (mt, cg): batch rows mt*16..+16, gate columns cg*16..+16 of r/z/n.
// Per step per wave: 3 gate tiles x 4 K-steps = 12 v_wmma.
// LDS: WhhB copy (96KB) + h fp32 (32KB) + h bf16 (16KB) = 144KB dynamic.
// ---------------------------------------------------------------------------
__global__ __launch_bounds__(1024, 1) void gru_scan(const __bf16* __restrict__ Whh,
                                                    const float* __restrict__ G,
                                                    const float* __restrict__ bhh,
                                                    float* __restrict__ Hout) {
    extern __shared__ char smem[];
    __bf16* whh_s = (__bf16*)smem;                              // 3H x H bf16
    float*  h_f   = (float*)(smem + (size_t)N3H * HH * 2);      // BS x H f32
    __bf16* h_b   = (__bf16*)(smem + (size_t)N3H * HH * 2 + (size_t)BS * HH * 4);

    const int tid  = threadIdx.x;
    const int wid  = tid >> 5;
    const int lane = tid & 31;
    const int half = lane >> 4;
    const int ln   = lane & 15;
    const int mt   = wid & 3;        // batch tile 0..3
    const int cg   = wid >> 2;       // column group 0..7

    // Preload Whh into LDS (98304 B = 6144 x uint4)
    for (int i = tid; i < (N3H * HH * 2) / 16; i += 1024)
        ((uint4*)whh_s)[i] = ((const uint4*)Whh)[i];
    // h0 = 0
    for (int i = tid; i < BS * HH; i += 1024) { h_f[i] = 0.0f; h_b[i] = (__bf16)0.0f; }
    __syncthreads();

    const int j = cg * 16 + ln;                 // hidden column this lane owns
    const float b_r = bhh[j];
    const float b_z = bhh[HH + j];
    const float b_n = bhh[2 * HH + j];

    const __bf16* arow = h_b + (size_t)(mt * 16 + ln) * HH + half * 16;
    const v8f vzero = {0.f, 0.f, 0.f, 0.f, 0.f, 0.f, 0.f, 0.f};

    for (int t = 0; t < SEQ; ++t) {
        v8f acc[3];
#pragma unroll
        for (int g = 0; g < 3; ++g) acc[g] = vzero;

#pragma unroll
        for (int kt = 0; kt < HH / 32; ++kt) {
            const int kk = kt * 32;
            v16bf a = load_a_frag(arow + kk);
#pragma unroll
            for (int g = 0; g < 3; ++g) {
                v16bf bfr = *(const v16bf*)(whh_s + (size_t)(g * HH + cg * 16 + ln) * HH + kk + half * 16);
                acc[g] = __builtin_amdgcn_wmma_f32_16x16x32_bf16(
                    false, a, false, bfr, (short)0, acc[g], false, false);
            }
        }

        // Gates entirely in registers; this wave owns r/z/n for its (m, j) patch.
        const float* gstep = G + (size_t)t * BS * N3H;
        if (t + 1 < SEQ)
            __builtin_prefetch(G + (size_t)(t + 1) * BS * N3H + (size_t)(mt * 16) * N3H + j, 0, 1);

        float hn[8];
#pragma unroll
        for (int r = 0; r < 8; ++r) {
            const int m = mt * 16 + r + 8 * half;
            const float* gi = gstep + (size_t)m * N3H;
            const float rr = sigmoidf_(gi[j]          + acc[0][r] + b_r);
            const float zz = sigmoidf_(gi[HH + j]     + acc[1][r] + b_z);
            const float nn = tanhf    (gi[2 * HH + j] + rr * (acc[2][r] + b_n));
            const float ho = h_f[(size_t)m * HH + j];
            hn[r] = (1.0f - zz) * nn + zz * ho;
        }
        __syncthreads();     // all reads of old h done
#pragma unroll
        for (int r = 0; r < 8; ++r) {
            const int m = mt * 16 + r + 8 * half;
            h_f[(size_t)m * HH + j] = hn[r];
            h_b[(size_t)m * HH + j] = (__bf16)hn[r];
        }
        __syncthreads();     // new h visible to everyone
    }

    for (int i = tid; i < BS * HH; i += 1024) Hout[i] = h_f[i];
}

// ---------------------------------------------------------------------------
// K4: backward-direction single cell on x[:, -1] from h0=0 (scalar, tiny)
// ---------------------------------------------------------------------------
__global__ void bwd_cell(const float* __restrict__ x, const float* __restrict__ mask,
                         const float* __restrict__ Wih, const float* __restrict__ bih,
                         const float* __restrict__ bhh, float* __restrict__ Hb) {
    const int gid = blockIdx.x * blockDim.x + threadIdx.x;
    if (gid >= BS * HH) return;
    const int m = gid >> 7;
    const int j = gid & (HH - 1);
    const size_t row = (size_t)m * SEQ + (SEQ - 1);
    const float mk = mask[row];
    const float* xr = x + row * EMB;
    float a0 = 0.f, a1 = 0.f, a2 = 0.f;
    const float* w0 = Wih + (size_t)j * EMB;
    const float* w1 = Wih + (size_t)(HH + j) * EMB;
    const float* w2 = Wih + (size_t)(2 * HH + j) * EMB;
    for (int k = 0; k < EMB; ++k) {
        const float xv = xr[k] * mk;
        a0 += xv * w0[k]; a1 += xv * w1[k]; a2 += xv * w2[k];
    }
    const float rr = sigmoidf_(a0 + bih[j] + bhh[j]);
    const float zz = sigmoidf_(a1 + bih[HH + j] + bhh[HH + j]);
    const float nn = tanhf(a2 + bih[2 * HH + j] + rr * bhh[2 * HH + j]);
    Hb[gid] = (1.0f - zz) * nn;          // z*h0 = 0
}

// ---------------------------------------------------------------------------
// K5: MLP head, one block per batch row. z = [h_fwd | h_bwd | s*s] (556)
// ---------------------------------------------------------------------------
__global__ __launch_bounds__(128) void mlp_head(const float* __restrict__ Hf,
                                                const float* __restrict__ Hb,
                                                const float* __restrict__ S,
                                                const float* __restrict__ W1, const float* __restrict__ b1,
                                                const float* __restrict__ W2, const float* __restrict__ b2,
                                                const float* __restrict__ W3, const float* __restrict__ b3,
                                                float* __restrict__ out) {
    const int m = blockIdx.x;
    const int tid = threadIdx.x;
    __shared__ float zv[2 * HH + EMB];   // 556
    __shared__ float z1[128];
    __shared__ float z2[64];

    for (int i = tid; i < 2 * HH + EMB; i += 128) {
        float v;
        if (i < HH)            v = Hf[(size_t)m * HH + i];
        else if (i < 2 * HH)   v = Hb[(size_t)m * HH + (i - HH)];
        else { float s = S[(size_t)m * EMB + (i - 2 * HH)]; v = s * s; }
        zv[i] = v;
    }
    __syncthreads();

    {   // layer 1: 556 -> 128, relu
        float acc = b1[tid];
        const float* w = W1 + (size_t)tid * (2 * HH + EMB);
        for (int k = 0; k < 2 * HH + EMB; ++k) acc += w[k] * zv[k];
        z1[tid] = fmaxf(acc, 0.0f);
    }
    __syncthreads();
    if (tid < 64) {   // layer 2: 128 -> 64, relu
        float acc = b2[tid];
        const float* w = W2 + (size_t)tid * 128;
        for (int k = 0; k < 128; ++k) acc += w[k] * z1[k];
        z2[tid] = fmaxf(acc, 0.0f);
    }
    __syncthreads();
    if (tid < NC) {   // layer 3: 64 -> 23
        float acc = b3[tid];
        const float* w = W3 + (size_t)tid * 64;
        for (int k = 0; k < 64; ++k) acc += w[k] * z2[k];
        out[(size_t)m * NC + tid] = acc;
    }
}

// ---------------------------------------------------------------------------
extern "C" void kernel_launch(void* const* d_in, const int* in_sizes, int n_in,
                              void* d_out, int out_size, void* d_ws, size_t ws_size,
                              hipStream_t stream) {
    (void)in_sizes; (void)n_in; (void)out_size; (void)ws_size;
    const float* x      = (const float*)d_in[0];
    /* d_in[1] = adj: dead code in the reference -> never touched (saves 268 MB) */
    const float* mask   = (const float*)d_in[2];
    const float* W_ih_f = (const float*)d_in[3];
    const float* W_hh_f = (const float*)d_in[4];
    const float* b_ih_f = (const float*)d_in[5];
    const float* b_hh_f = (const float*)d_in[6];
    const float* W_ih_b = (const float*)d_in[7];
    /* d_in[8] = W_hh_b unused: h0 = 0 for the single backward step */
    const float* b_ih_b = (const float*)d_in[9];
    const float* b_hh_b = (const float*)d_in[10];
    const float* W1 = (const float*)d_in[11];
    const float* b1 = (const float*)d_in[12];
    const float* W2 = (const float*)d_in[13];
    const float* b2 = (const float*)d_in[14];
    const float* W3 = (const float*)d_in[15];
    const float* b3 = (const float*)d_in[16];
    float* out = (float*)d_out;

    char* ws = (char*)d_ws;
    __bf16* XM   = (__bf16*)(ws + WS_XM);
    float*  G    = (float*) (ws + WS_G);
    __bf16* WIH  = (__bf16*)(ws + WS_WIH);
    __bf16* WHH  = (__bf16*)(ws + WS_WHH);
    float*  S    = (float*) (ws + WS_S);
    float*  HF   = (float*) (ws + WS_HF);
    float*  HB   = (float*) (ws + WS_HB);

    // K0: weights -> bf16
    {
        const int total = N3H * KPAD + N3H * HH;
        prep_weights<<<(total + 255) / 256, 256, 0, stream>>>(W_ih_f, W_hh_f, WIH, WHH);
    }
    // K1: mask + convert + feature-sum
    mask_convert_sum<<<BS, KPAD, 0, stream>>>(x, mask, XM, S);
    // K2: input-projection WMMA GEMM, time-major output
    {
        dim3 grid((BS * SEQ) / 128, N3H / 64);
        gemm_gi<<<grid, 256, 0, stream>>>(XM, WIH, b_ih_f, G);
    }
    // K3: sequential scan (single WGP, 147456 B dynamic LDS)
    {
        const int lds = N3H * HH * 2 + BS * HH * 4 + BS * HH * 2;   // 147456
        (void)hipFuncSetAttribute((const void*)gru_scan,
                                  hipFuncAttributeMaxDynamicSharedMemorySize, lds);
        gru_scan<<<1, 1024, lds, stream>>>(WHH, G, b_hh_f, HF);
    }
    // K4: backward single cell
    bwd_cell<<<(BS * HH + 255) / 256, 256, 0, stream>>>(x, mask, W_ih_b, b_ih_b, b_hh_b, HB);
    // K5: MLP head -> logits
    mlp_head<<<BS, 128, 0, stream>>>(HF, HB, S, W1, b1, W2, b2, W3, b3, out);
}
